// Attention_43516608643501
// MI455X (gfx1250) — matrix operand order repaired and checked by reading, not compiled, not run on hardware
//
#include <hip/hip_runtime.h>

// ---------------------------------------------------------------------------
// Flash-attention for: scores = od·oe^T (per batch), softmax over SRC dim,
// context = P·oe.  oe = out_e[...,:512] + out_e[...,512:].
// SRC=TGT=2048, B=16, H=512.  Output [TGT, B, H] fp32.
// bf16 WMMA (v_wmma_f32_16x16x32_bf16), f32 accumulate, online softmax.
// v3: 512-thread blocks (16 waves), TBLK=128 -> 4 waves/SIMD from one
//     workgroup (294 KB LDS of the WGP's 320 KB), 2x K-tile reuse and half
//     the L2 re-read traffic vs the 256-thread version.
// ---------------------------------------------------------------------------

#define SRC   2048
#define TGT   2048
#define NB    16
#define HH    512
#define HD2   1024
#define TBLK  128         // t-rows per workgroup
#define SBLK  64          // s-rows per iteration
#define NTHR  512         // 16 waves
#define QSTR  520         // row stride (bf16) for Qs/Ks  (512 + 8 pad)
#define TSTR  72          // row stride (bf16) for Kt/Ps  (64 + 8 pad)
#define L2E   1.4426950408889634f

typedef __bf16 bf16;
typedef __attribute__((ext_vector_type(16))) __bf16 v16bf;
typedef __attribute__((ext_vector_type(8)))  __bf16 v8bf;
typedef __attribute__((ext_vector_type(8)))  float  v8f;

static __device__ __forceinline__ v16bf cat16(v8bf lo, v8bf hi) {
    return __builtin_shufflevector(lo, hi, 0,1,2,3,4,5,6,7,8,9,10,11,12,13,14,15);
}
// B operand: 16 contiguous bf16 (lane n = column, K ascending)
static __device__ __forceinline__ v16bf ldB(const bf16* p) {
    return cat16(*(const v8bf*)p, *(const v8bf*)(p + 8));
}
// A operand: 16 contiguous bf16 (lane m = row, K ascending)
static __device__ __forceinline__ v16bf ldA(const bf16* p) {
    return cat16(*(const v8bf*)p, *(const v8bf*)(p + 8));
}
static __device__ __forceinline__ float rmax16(float v) {
    v = fmaxf(v, __shfl_xor(v, 1, 32));
    v = fmaxf(v, __shfl_xor(v, 2, 32));
    v = fmaxf(v, __shfl_xor(v, 4, 32));
    v = fmaxf(v, __shfl_xor(v, 8, 32));
    return v;
}
static __device__ __forceinline__ float rsum16(float v) {
    v += __shfl_xor(v, 1, 32);
    v += __shfl_xor(v, 2, 32);
    v += __shfl_xor(v, 4, 32);
    v += __shfl_xor(v, 8, 32);
    return v;
}

extern __shared__ char smem_raw[];

__global__ __launch_bounds__(NTHR)
void attn_flash_kernel(const float* __restrict__ out_e,
                       const float* __restrict__ out_d,
                       float* __restrict__ out) {
    bf16*  Qs   = (bf16*)smem_raw;            // [128][520]
    bf16*  Ks   = Qs + TBLK * QSTR;           // [64][520]  row-major K/V
    bf16*  Kt   = Ks + SBLK * QSTR;           // [512][72]  transposed V
    bf16*  Ps   = Kt + HH * TSTR;             // [128][72]  probabilities
    float* pmax = (float*)(Ps + TBLK * TSTR); // [2][128]
    float* psum = pmax + 2 * TBLK;            // [2][128]

    const int tid  = threadIdx.x;
    const int wave = tid >> 5;    // 0..15
    const int lane = tid & 31;
    const int ln   = lane & 15;
    const int lh   = lane >> 4;
    const int tt   = wave >> 1;   // t-tile (0..7)
    const int half = wave & 1;    // s-half for scores / h-half for PV
    const int b    = blockIdx.x & (NB - 1);
    const int t0   = (blockIdx.x >> 4) * TBLK;

    // ---- stage Q tile: out_d[t0..t0+127, b, 0..511] -> bf16 LDS ------------
    for (int i = tid; i < TBLK * HH / 4; i += NTHR) {
        int row = i >> 7;
        int h4  = (i & 127) << 2;
        const float4 q = *(const float4*)(out_d + (size_t)(t0 + row) * (NB * HH)
                                          + (size_t)b * HH + h4);
        union { bf16 h[4]; unsigned long long u; } pk;
        pk.h[0] = (bf16)q.x; pk.h[1] = (bf16)q.y;
        pk.h[2] = (bf16)q.z; pk.h[3] = (bf16)q.w;
        *(unsigned long long*)(Qs + row * QSTR + h4) = pk.u;
    }

    const v8f vzero = {0.f, 0.f, 0.f, 0.f, 0.f, 0.f, 0.f, 0.f};
    v8f acc[16];                       // 16t x 256h fp32 accumulators
#pragma unroll
    for (int n = 0; n < 16; ++n) acc[n] = vzero;
    float mrow[8], lrow[8];
#pragma unroll
    for (int r = 0; r < 8; ++r) { mrow[r] = -__builtin_inff(); lrow[r] = 0.f; }

    for (int it = 0; it < SRC / SBLK; ++it) {
        const int s0 = it * SBLK;

        // ---- stage K/V tile: oe = fwd+bwd halves, bf16, two layouts -------
        for (int i = tid; i < SBLK * HH / 4; i += NTHR) {
            int row = i >> 7;
            int h4  = (i & 127) << 2;
            const float* g = out_e + (size_t)(s0 + row) * (NB * HD2)
                             + (size_t)b * HD2 + h4;
            const float4 e0 = *(const float4*)g;
            const float4 e1 = *(const float4*)(g + HH);
            union { bf16 h[4]; unsigned long long u; } pk;
            pk.h[0] = (bf16)(e0.x + e1.x); pk.h[1] = (bf16)(e0.y + e1.y);
            pk.h[2] = (bf16)(e0.z + e1.z); pk.h[3] = (bf16)(e0.w + e1.w);
            *(unsigned long long*)(Ks + row * QSTR + h4) = pk.u;
            Kt[(h4 + 0) * TSTR + row] = pk.h[0];
            Kt[(h4 + 1) * TSTR + row] = pk.h[1];
            Kt[(h4 + 2) * TSTR + row] = pk.h[2];
            Kt[(h4 + 3) * TSTR + row] = pk.h[3];
        }
        __syncthreads();

        // prefetch next K tile into cache while we compute (global_prefetch_b8)
        if (it + 1 < SRC / SBLK) {
            const float* nxt = out_e + (size_t)(s0 + SBLK + (tid >> 3)) * (NB * HD2)
                               + (size_t)b * HD2 + ((tid & 7) << 6);
            __builtin_prefetch(nxt, 0, 0);
        }

        // ---- scores: 16t x 32s per wave, K-depth 512 ----------------------
        v8f c0 = vzero, c1 = vzero;
        const bf16* qrow = Qs + (tt * 16 + ln) * QSTR + lh * 16;
        const bf16* k0   = Ks + (half * 32 + ln) * QSTR + lh * 16;
        const bf16* k1   = k0 + 16 * QSTR;
        v16bf a  = ldA(qrow);
        v16bf b0 = ldB(k0);
        v16bf b1 = ldB(k1);
#pragma unroll
        for (int k = 0; k < 16; ++k) {
            v16bf an = a, b0n = b0, b1n = b1;
            if (k < 15) {
                an  = ldA(qrow + (k + 1) * 32);
                b0n = ldB(k0 + (k + 1) * 32);
                b1n = ldB(k1 + (k + 1) * 32);
            }
            c0 = __builtin_amdgcn_wmma_f32_16x16x32_bf16(false, a, false, b0,
                                                         (short)0, c0, false, false);
            c1 = __builtin_amdgcn_wmma_f32_16x16x32_bf16(false, a, false, b1,
                                                         (short)0, c1, false, false);
            a = an; b0 = b0n; b1 = b1n;
        }

        // ---- online softmax over s ----------------------------------------
        float pm[8];
#pragma unroll
        for (int r = 0; r < 8; ++r) pm[r] = rmax16(fmaxf(c0[r], c1[r]));
        if (ln == 0) {
#pragma unroll
            for (int r = 0; r < 8; ++r)
                pmax[half * TBLK + tt * 16 + lh * 8 + r] = pm[r];
        }
        __syncthreads();

        float sc[8], mn[8];
#pragma unroll
        for (int r = 0; r < 8; ++r) {
            const int row = tt * 16 + lh * 8 + r;
            const float m2 = fmaxf(pmax[row], pmax[TBLK + row]);
            mn[r]  = fmaxf(mrow[r], m2);
            sc[r]  = exp2f((mrow[r] - mn[r]) * L2E);
            mrow[r] = mn[r];
        }
        float ps[8];
#pragma unroll
        for (int r = 0; r < 8; ++r) {
            const float p0 = exp2f((c0[r] - mn[r]) * L2E);
            const float p1 = exp2f((c1[r] - mn[r]) * L2E);
            ps[r] = rsum16(p0 + p1);
            const int row = tt * 16 + lh * 8 + r;
            Ps[row * TSTR + half * 32 + ln]      = (bf16)p0;
            Ps[row * TSTR + half * 32 + 16 + ln] = (bf16)p1;
        }
        if (ln == 0) {
#pragma unroll
            for (int r = 0; r < 8; ++r)
                psum[half * TBLK + tt * 16 + lh * 8 + r] = ps[r];
        }
        // rescale accumulators (per-row factor, C-layout rows = VGPR index)
#pragma unroll
        for (int n = 0; n < 16; ++n)
#pragma unroll
            for (int r = 0; r < 8; ++r) acc[n][r] *= sc[r];
        __syncthreads();
#pragma unroll
        for (int r = 0; r < 8; ++r) {
            const int row = tt * 16 + lh * 8 + r;
            lrow[r] = lrow[r] * sc[r] + psum[row] + psum[TBLK + row];
        }

        // ---- PV: acc += P(16t x 64s) * V(64s x 256h) ----------------------
        const int h0 = half * 256;
#pragma unroll
        for (int kk = 0; kk < 2; ++kk) {
            const bf16* vbase = Kt + (h0 + ln) * TSTR + kk * 32 + lh * 16;
            v16bf av = ldA(Ps + (tt * 16 + ln) * TSTR + kk * 32 + lh * 16);
            v16bf bv0 = ldB(vbase);
            v16bf bv1 = ldB(vbase + 16 * TSTR);
#pragma unroll
            for (int n = 0; n < 16; ++n) {
                v16bf bnext = bv1;
                if (n < 14) bnext = ldB(vbase + (n + 2) * 16 * TSTR);
                acc[n] = __builtin_amdgcn_wmma_f32_16x16x32_bf16(false, av, false, bv0,
                                                                 (short)0, acc[n],
                                                                 false, false);
                bv0 = bv1; bv1 = bnext;
            }
        }
        __syncthreads();
    }

    // ---- normalize and store: out[t, b, h] --------------------------------
    float inv[8];
#pragma unroll
    for (int r = 0; r < 8; ++r) inv[r] = 1.0f / lrow[r];
#pragma unroll
    for (int n = 0; n < 16; ++n) {
        const int h = half * 256 + n * 16 + ln;
#pragma unroll
        for (int r = 0; r < 8; ++r) {
            const int t = t0 + tt * 16 + lh * 8 + r;
            out[(size_t)t * (NB * HH) + (size_t)b * HH + h] = acc[n][r] * inv[r];
        }
    }
}

extern "C" void kernel_launch(void* const* d_in, const int* in_sizes, int n_in,
                              void* d_out, int out_size, void* d_ws, size_t ws_size,
                              hipStream_t stream) {
    (void)in_sizes; (void)n_in; (void)out_size; (void)d_ws; (void)ws_size;
    // setup_inputs order: in_e (int, unused), out_e (f32), out_d (f32)
    const float* out_e = (const float*)d_in[1];
    const float* out_d = (const float*)d_in[2];
    float* out = (float*)d_out;

    const size_t shmem =
        (size_t)(TBLK * QSTR + SBLK * QSTR + HH * TSTR + TBLK * TSTR) * sizeof(bf16)
        + (size_t)4 * TBLK * sizeof(float);   // 293,888 bytes < 320 KB WGP LDS

    dim3 grid((TGT / TBLK) * NB);   // 16 * 16 = 256 workgroups
    hipLaunchKernelGGL(attn_flash_kernel, grid, dim3(NTHR), shmem, stream,
                       out_e, out_d, out);
}
